// TemporalCausalAttention_55130200211531
// MI455X (gfx1250) — matrix operand order repaired and checked by reading, not compiled, not run on hardware
//
#include <hip/hip_runtime.h>
#include <hip/hip_bf16.h>

typedef __bf16 bf16_t;
typedef __bf16 v8bf  __attribute__((ext_vector_type(8)));
typedef __bf16 v16bf __attribute__((ext_vector_type(16)));
typedef float  v8f   __attribute__((ext_vector_type(8)));

// Problem constants (B=2, T=64, S=1024, D=512, H=8, hd=64)
#define QK_STRIDE 1032   // 64 rows x (1024 q,k cols + pad) bf16
#define VT_STRIDE 72     // 512 rows (h*64+d) x (64 t cols + pad) bf16, V stored transposed
#define XS_STRIDE 520    // 64 rows x (512 cols + pad) bf16 (x slice, reused for attn-out)
#define PS_STRIDE 72     // per-wave P scratch: 16 rows x (64 + pad)
#define SMEM_BF16 (64*QK_STRIDE + 512*VT_STRIDE + 64*XS_STRIDE + 8*16*PS_STRIDE)
#define SMEM_BYTES (SMEM_BF16 * 2)

#define NQ_PACK (96*16*512)   // Wqkv packed bf16 elements (96 n-tiles x 16 k-tiles x 512)
#define NO_PACK (32*16*512)   // Wout packed bf16 elements

static __device__ __forceinline__ v8f wmma_bf16(v16bf a, v16bf b, v8f c) {
  // D = A(16x32) * B(32x16) + C, f32 accumulate
  return __builtin_amdgcn_wmma_f32_16x16x32_bf16(false, a, false, b, (short)0, c, false, false);
}

// A-fragment (16x32 bf16) from row-major [M][K] LDS buffer.
// Contiguous k order used here: lanes 0-15: M=lane, K = kb+{0..15};
//             lanes 16-31: M=lane-16, K = kb+16+{0..15}.
static __device__ __forceinline__ v16bf load_afrag(const bf16_t* base, int stride,
                                                   int row_base, int kbase) {
  const int l  = threadIdx.x & 31;
  const int hi = l >> 4;
  const bf16_t* p0 = base + (row_base + (l & 15)) * stride + kbase + hi * 16;
  v8bf lo = *(const v8bf*)p0;
  v8bf up = *(const v8bf*)(p0 + 8);
  return __builtin_shufflevector(lo, up, 0,1,2,3,4,5,6,7,8,9,10,11,12,13,14,15);
}

// B-fragment (32x16 bf16) from row-major [N][K] LDS buffer.
// Contiguous k order used here: lanes 0-15: N=lane, K = kb+{0..15}; lanes 16-31: N=lane-16, K = kb+16+{0..15}.
static __device__ __forceinline__ v16bf load_bfrag(const bf16_t* base, int stride,
                                                   int row_base, int kbase) {
  const int l  = threadIdx.x & 31;
  const int hi = l >> 4;
  const bf16_t* p0 = base + (row_base + (l & 15)) * stride + kbase + hi * 16;
  v8bf lo = *(const v8bf*)p0;
  v8bf up = *(const v8bf*)(p0 + 8);
  return __builtin_shufflevector(lo, up, 0,1,2,3,4,5,6,7,8,9,10,11,12,13,14,15);
}

// Kernel 0: pack Wqkv [512,1536] and Wout [512,512] fp32 -> bf16 B-fragment layout.
// Packed element index: ((nt*16 + kt)*512) + lane*16 + j, where
// n = nt*16 + (lane&15), k = kt*32 + (lane>>4)*16 + j.
__global__ void pack_weights(const float* __restrict__ Wqkv, const float* __restrict__ Wout,
                             bf16_t* __restrict__ pQKV, bf16_t* __restrict__ pWout) {
  int idx = blockIdx.x * 256 + threadIdx.x;
  if (idx < NQ_PACK) {
    int j    = idx & 15;
    int lane = (idx >> 4) & 31;
    int kt   = (idx >> 9) & 15;
    int nt   = idx >> 13;
    int k = kt * 32 + (lane >> 4) * 16 + j;
    int n = nt * 16 + (lane & 15);
    pQKV[idx] = (bf16_t)Wqkv[k * 1536 + n];
  } else if (idx < NQ_PACK + NO_PACK) {
    int i2   = idx - NQ_PACK;
    int j    = i2 & 15;
    int lane = (i2 >> 4) & 31;
    int kt   = (i2 >> 9) & 15;
    int nt   = i2 >> 13;
    int k = kt * 32 + (lane >> 4) * 16 + j;
    int n = nt * 16 + (lane & 15);
    pWout[i2] = (bf16_t)Wout[k * 512 + n];
  }
}

// Fused: x->qkv (GEMM1), causal attention per head, out proj (GEMM2).
// One block per (b,s); 8 waves; wave w owns head w in the attention phase.
__global__ __launch_bounds__(256)
void attn_fused(const float* __restrict__ x, const bf16_t* __restrict__ pQKV,
                const bf16_t* __restrict__ pWout, const float* __restrict__ bout,
                float* __restrict__ out) {
  extern __shared__ __align__(16) char smem_raw[];
  bf16_t* qk = (bf16_t*)smem_raw;            // [64][QK_STRIDE] : q (n<512), k (512..1023)
  bf16_t* vT = qk + 64 * QK_STRIDE;          // [512][VT_STRIDE]: v transposed (row = h*64+d, col = t)
  bf16_t* xs = vT + 512 * VT_STRIDE;         // [64][XS_STRIDE] : x slice, later attn output
  bf16_t* ps = xs + 64 * XS_STRIDE;          // [8][16][PS_STRIDE]: per-wave P scratch

  const int tid  = threadIdx.x;
  const int lane = tid & 31;
  const int w    = tid >> 5;
  const int lsub = lane & 15;
  const int hi   = lane >> 4;
  const int bs   = blockIdx.x;
  const int b    = bs >> 10;
  const int s    = bs & 1023;

  // ---- Phase 0: load x[b, :, s, :] (64x512 f32) -> bf16 LDS ----
  for (int i = tid; i < 64 * 512; i += 256) {
    int t = i >> 9, d = i & 511;
    xs[t * XS_STRIDE + d] = (bf16_t)x[((b * 64 + t) * 1024 + s) * 512 + d];
  }
  __syncthreads();

  // ---- Phase 1: GEMM1 qkv[64,1536] = x @ Wqkv ----
  // 12 n-tiles per wave, register-blocked in pairs: each A-frag feeds 2 WMMAs.
  for (int jp = 0; jp < 6; ++jp) {
    int nt0 = w * 12 + jp * 2;
    const bf16_t* wb0 = pQKV + (size_t)nt0 * (16 * 512) + lane * 16;
    const bf16_t* wb1 = wb0 + 16 * 512;
    v8f acc0[4] = {}, acc1[4] = {};
    for (int kt = 0; kt < 16; ++kt) {
      if (kt + 1 < 16) {
        __builtin_prefetch(wb0 + (kt + 1) * 512, 0, 3);   // near (WGP) scope
        __builtin_prefetch(wb1 + (kt + 1) * 512, 0, 3);
      }
      v16bf bfr0 = *(const v16bf*)(wb0 + kt * 512);
      v16bf bfr1 = *(const v16bf*)(wb1 + kt * 512);
      #pragma unroll
      for (int mt = 0; mt < 4; ++mt) {
        v16bf afr = load_afrag(xs, XS_STRIDE, mt * 16, kt * 32);
        acc0[mt] = wmma_bf16(afr, bfr0, acc0[mt]);
        acc1[mt] = wmma_bf16(afr, bfr1, acc1[mt]);
      }
    }
    #pragma unroll
    for (int half = 0; half < 2; ++half) {
      int ng = (nt0 + half) * 16 + lsub;   // qkv column 0..1535 (uniform region per nt)
      v8f* acc = half ? acc1 : acc0;
      #pragma unroll
      for (int mt = 0; mt < 4; ++mt) {
        #pragma unroll
        for (int r = 0; r < 8; ++r) {
          int t = mt * 16 + r + hi * 8;    // C layout: VGPR r, lanes 0-15 -> M=r, 16-31 -> M=r+8
          bf16_t v = (bf16_t)acc[mt][r];
          if (ng < 1024) qk[t * QK_STRIDE + ng] = v;           // q, k row-major
          else           vT[(ng - 1024) * VT_STRIDE + t] = v;  // v transposed
        }
      }
    }
  }
  __syncthreads();

  // ---- Phase 2: causal attention, head h = wave id ----
  const int h = w;
  bf16_t* myp = ps + w * 16 * PS_STRIDE;

  // K and V fragments are invariant across query blocks: hoist into registers.
  v16bf kfr[2][4], vfr[2][4];
  #pragma unroll
  for (int kt = 0; kt < 2; ++kt) {
    #pragma unroll
    for (int ntb = 0; ntb < 4; ++ntb) {
      kfr[kt][ntb] = load_bfrag(qk, QK_STRIDE, ntb * 16, 512 + h * 64 + kt * 32);
      vfr[kt][ntb] = load_bfrag(vT, VT_STRIDE, h * 64 + ntb * 16, kt * 32);
    }
  }

  for (int mt = 0; mt < 4; ++mt) {
    // scores S[16,64] = q_block @ k^T
    v8f sc[4] = {};
    #pragma unroll
    for (int kt = 0; kt < 2; ++kt) {
      v16bf qa = load_afrag(qk, QK_STRIDE, mt * 16, h * 64 + kt * 32);
      #pragma unroll
      for (int ntb = 0; ntb < 4; ++ntb)
        sc[ntb] = wmma_bf16(qa, kfr[kt][ntb], sc[ntb]);
    }
    // scale + causal mask + row-max (rows live in 16-lane groups; reduce with shfl_xor)
    float rmax[8], rsum[8];
    #pragma unroll
    for (int r = 0; r < 8; ++r) rmax[r] = -1e30f;
    #pragma unroll
    for (int ntb = 0; ntb < 4; ++ntb) {
      #pragma unroll
      for (int r = 0; r < 8; ++r) {
        int tq = mt * 16 + r + hi * 8;
        int tk = ntb * 16 + lsub;
        float v = sc[ntb][r] * 0.125f;              // hd^-0.5
        v = (tk <= tq) ? v : -1e30f;
        sc[ntb][r] = v;
        rmax[r] = fmaxf(rmax[r], v);
      }
    }
    #pragma unroll
    for (int r = 0; r < 8; ++r) {
      #pragma unroll
      for (int m = 1; m < 16; m <<= 1)
        rmax[r] = fmaxf(rmax[r], __shfl_xor(rmax[r], m, 32));
      rsum[r] = 0.f;
    }
    #pragma unroll
    for (int ntb = 0; ntb < 4; ++ntb) {
      #pragma unroll
      for (int r = 0; r < 8; ++r) {
        float e = __expf(sc[ntb][r] - rmax[r]);
        sc[ntb][r] = e;
        rsum[r] += e;
      }
    }
    #pragma unroll
    for (int r = 0; r < 8; ++r) {
      #pragma unroll
      for (int m = 1; m < 16; m <<= 1)
        rsum[r] += __shfl_xor(rsum[r], m, 32);
      rsum[r] = 1.0f / rsum[r];
    }
    // P -> bf16 in wave-private LDS scratch (rows 0..15, cols = t_key)
    #pragma unroll
    for (int ntb = 0; ntb < 4; ++ntb) {
      #pragma unroll
      for (int r = 0; r < 8; ++r)
        myp[(r + hi * 8) * PS_STRIDE + ntb * 16 + lsub] = (bf16_t)(sc[ntb][r] * rsum[r]);
    }
    // O[16,64] = P @ V  (B-frags contiguous thanks to transposed V)
    v8f oa[4] = {};
    #pragma unroll
    for (int kt = 0; kt < 2; ++kt) {
      v16bf pa = load_afrag(myp, PS_STRIDE, 0, kt * 32);
      #pragma unroll
      for (int ntd = 0; ntd < 4; ++ntd)
        oa[ntd] = wmma_bf16(pa, vfr[kt][ntd], oa[ntd]);
    }
    // store attn output rows into xs (reused as [64,512] attn buffer)
    #pragma unroll
    for (int ntd = 0; ntd < 4; ++ntd) {
      #pragma unroll
      for (int r = 0; r < 8; ++r)
        xs[(mt * 16 + r + hi * 8) * XS_STRIDE + h * 64 + ntd * 16 + lsub] = (bf16_t)oa[ntd][r];
    }
  }
  __syncthreads();

  // ---- Phase 3: GEMM2 out[64,512] = attn @ Wout + bout ----
  // 4 n-tiles per wave, register-blocked in pairs.
  for (int jp = 0; jp < 2; ++jp) {
    int nt0 = w * 4 + jp * 2;
    const bf16_t* wb0 = pWout + (size_t)nt0 * (16 * 512) + lane * 16;
    const bf16_t* wb1 = wb0 + 16 * 512;
    v8f acc0[4] = {}, acc1[4] = {};
    for (int kt = 0; kt < 16; ++kt) {
      if (kt + 1 < 16) {
        __builtin_prefetch(wb0 + (kt + 1) * 512, 0, 3);
        __builtin_prefetch(wb1 + (kt + 1) * 512, 0, 3);
      }
      v16bf bfr0 = *(const v16bf*)(wb0 + kt * 512);
      v16bf bfr1 = *(const v16bf*)(wb1 + kt * 512);
      #pragma unroll
      for (int mt = 0; mt < 4; ++mt) {
        v16bf afr = load_afrag(xs, XS_STRIDE, mt * 16, kt * 32);
        acc0[mt] = wmma_bf16(afr, bfr0, acc0[mt]);
        acc1[mt] = wmma_bf16(afr, bfr1, acc1[mt]);
      }
    }
    #pragma unroll
    for (int half = 0; half < 2; ++half) {
      int n = (nt0 + half) * 16 + lsub;
      float bias = bout[n];
      v8f* acc = half ? acc1 : acc0;
      #pragma unroll
      for (int mt = 0; mt < 4; ++mt) {
        #pragma unroll
        for (int r = 0; r < 8; ++r) {
          int t = mt * 16 + r + hi * 8;
          out[((b * 64 + t) * 1024 + s) * 512 + n] = acc[mt][r] + bias;
        }
      }
    }
  }
}

extern "C" void kernel_launch(void* const* d_in, const int* in_sizes, int n_in,
                              void* d_out, int out_size, void* d_ws, size_t ws_size,
                              hipStream_t stream) {
  const float* x    = (const float*)d_in[0];
  const float* Wqkv = (const float*)d_in[1];
  const float* Wout = (const float*)d_in[2];
  const float* bout = (const float*)d_in[3];
  float* out = (float*)d_out;

  bf16_t* pQKV  = (bf16_t*)d_ws;             // 1.5 MB
  bf16_t* pWout = pQKV + NQ_PACK;            // 0.5 MB

  static_assert(SMEM_BYTES <= 320 * 1024, "LDS over budget");
  (void)hipFuncSetAttribute(reinterpret_cast<const void*>(&attn_fused),
                            hipFuncAttributeMaxDynamicSharedMemorySize, SMEM_BYTES);

  int pack_threads = NQ_PACK + NO_PACK;
  pack_weights<<<(pack_threads + 255) / 256, 256, 0, stream>>>(Wqkv, Wout, pQKV, pWout);
  attn_fused<<<2048, 256, SMEM_BYTES, stream>>>(x, pQKV, pWout, bout, out);
}
